// test_Sparse_UNet_86457691668780
// MI455X (gfx1250) — matrix operand order repaired and checked by reading, not compiled, not run on hardware
//
#include <hip/hip_runtime.h>
#include <stdint.h>

typedef __attribute__((ext_vector_type(16))) _Float16 v16h;
typedef __attribute__((ext_vector_type(8)))  _Float16 v8h;
typedef __attribute__((ext_vector_type(8)))  float    v8f;

#define EPS_BN 1e-4f
#define KNBR 27

// ---------------------------------------------------------------------------
// zero a float region of workspace
// ---------------------------------------------------------------------------
__global__ void zero_f32(float* __restrict__ p, int n) {
    int i = blockIdx.x * blockDim.x + threadIdx.x;
    if (i < n) p[i] = 0.0f;
}

// ---------------------------------------------------------------------------
// Pack fp32 features -> channel-padded f16, with optional fused BN+ReLU.
//   xh[n][c] = c < Cin ? relu(x[n][c]*scale[c]+shift[c]) : 0
// ---------------------------------------------------------------------------
template<int Cin, int CinPad, bool APPLY_BN>
__global__ __launch_bounds__(256)
void pack_f16(const float* __restrict__ x, const float* __restrict__ scale,
              const float* __restrict__ shift, _Float16* __restrict__ xh, int N) {
    int i = blockIdx.x * blockDim.x + threadIdx.x;
    if (i < N * CinPad) {
        int c = i & (CinPad - 1);
        float v = 0.0f;
        if (c < Cin) {
            v = x[(long long)(i >> __builtin_ctz(CinPad)) * Cin + c];
            if (APPLY_BN) v = fmaxf(fmaf(v, scale[c], shift[c]), 0.0f);
        }
        xh[i] = (_Float16)v;
    }
}

// ---------------------------------------------------------------------------
// Pack weights [27, Cin, Cout] fp32 -> WMMA B-fragment-ordered f16 buffer:
//   Bp[(chunk*COLT + ct)*512 + lane*16 + i] = Wflat(chunk*32 + 16*(i/8) + 8*hi + i%8,
//                                                   ct*16 + (lane&15))
// with zeros in all K padding (koff >= 27 or c >= Cin). Run once per layer.
// ---------------------------------------------------------------------------
template<int Cin, int CinPad, int Cout, int KPad>
__global__ __launch_bounds__(256)
void pack_B(const float* __restrict__ W, _Float16* __restrict__ Bp) {
    constexpr int NCHUNK = KPad / 32;
    constexpr int COLT   = Cout / 16;
    constexpr int CSHIFT = (CinPad == 8) ? 3 : (CinPad == 16) ? 4 : 5;
    int idx = blockIdx.x * blockDim.x + threadIdx.x;
    if (idx < NCHUNK * COLT * 512) {
        int comb  = idx >> 9;                   // chunk*COLT + ct
        int rem   = idx & 511;
        int ct    = (COLT > 1) ? (comb % COLT) : 0;
        int chunk = (COLT > 1) ? (comb / COLT) : comb;
        int lane  = rem >> 4;
        int i     = rem & 15;
        int kk    = chunk * 32 + ((i >> 3) << 4) + ((lane >> 4) << 3) + (i & 7);   // kb + 16*(i/8) + 8*hi + i%8
        int koff  = kk >> CSHIFT;
        int c     = kk & (CinPad - 1);
        float v = 0.0f;
        if (koff < KNBR && c < Cin)
            v = W[(koff * Cin + c) * Cout + ct * 16 + (lane & 15)];
        Bp[idx] = (_Float16)v;
    }
}

// ---------------------------------------------------------------------------
// Submanifold conv as flattened GEMM, one wave per 16-voxel x 16-col tile.
//   out[N,Cout] = G[N, 27*CinPad] @ Wflat[27*CinPad, Cout]
// A: two aligned global_load_b128 (f16) per lane per 32-K chunk, no branches.
// B: pre-packed fragments async-copied global -> LDS (ASYNCcnt path).
// Accumulates per-channel sum / sum-of-squares for BatchNorm stats.
// ---------------------------------------------------------------------------
template<int CinPad, int Cout, int KPad>
__global__ __launch_bounds__(128)
void conv_wmma(const _Float16* __restrict__ xh,   // [N, CinPad] packed f16
               const int*      __restrict__ nbr,  // [N, 27]
               const _Float16* __restrict__ Bp,   // fragment-ordered weights
               float* __restrict__ hout,          // [N, Cout] raw conv out
               float* __restrict__ sumOut,        // [Cout]
               float* __restrict__ sqOut,         // [Cout]
               int N) {
    constexpr int NCHUNK = KPad / 32;
    constexpr int COLT   = Cout / 16;
    constexpr int WAVES  = 4;
    constexpr int CSHIFT = (CinPad == 8) ? 3 : (CinPad == 16) ? 4 : 5;
    constexpr int NVEC   = NCHUNK * COLT * 64;    // 16B vectors to stage

    __shared__ _Float16 Bsm[NCHUNK * COLT * 512]; // [(chunk*COLT+ct)][lane][i]

    const int tid = threadIdx.x;

    // ---- stage pre-packed B fragments: async global -> LDS, 16B per op ----
    for (int idx = tid; idx < NVEC; idx += 128) {
        unsigned lds            = (unsigned)(uintptr_t)&Bsm[idx * 8];
        const _Float16* gsrc    = Bp + idx * 8;
        asm volatile("global_load_async_to_lds_b128 %0, %1, off"
                     :: "v"(lds), "v"(gsrc)
                     : "memory");
    }
    asm volatile("s_wait_asynccnt 0x0" ::: "memory");
    __syncthreads();

    const int wave = tid >> 5;
    const int lane = tid & 31;
    const int hi   = lane >> 4;
    const int nloc = lane & 15;

    const int gw      = blockIdx.x * WAVES + wave;
    const int tileId  = gw / COLT;
    const int ct      = gw % COLT;
    const int rowBase = tileId * 16;
    if (rowBase >= N) return;                 // wave-uniform guard

    const int row = rowBase + nloc;           // A row for this lane (both halves)
    const int* __restrict__ nbrRow = nbr + (long long)row * KNBR;

    v8f acc = {};

    for (int chunk = 0; chunk < NCHUNK; ++chunk) {
        // ---- A fragment: two aligned 16B f16 loads ----
        int k0  = chunk * 32 + hi * 8;
        int k1  = k0 + 16;
        int ko0 = k0 >> CSHIFT;  if (ko0 >= KNBR) ko0 = KNBR - 1;  // tail: B is 0
        int ko1 = k1 >> CSHIFT;  if (ko1 >= KNBR) ko1 = KNBR - 1;
        int c0  = k0 & (CinPad - 1);
        int c1  = k1 & (CinPad - 1);
        int nb0 = nbrRow[ko0];
        int nb1 = nbrRow[ko1];
        v8h a0 = *reinterpret_cast<const v8h*>(xh + ((long long)nb0 << CSHIFT) + c0);
        v8h a1 = *reinterpret_cast<const v8h*>(xh + ((long long)nb1 << CSHIFT) + c1);
        v16h a = __builtin_shufflevector(a0, a1,
                                         0, 1, 2, 3, 4, 5, 6, 7,
                                         8, 9, 10, 11, 12, 13, 14, 15);
        // ---- B fragment: two contiguous 16B LDS reads ----
        const _Float16* bp = &Bsm[(((chunk * COLT + ct) * 32) + lane) * 16];
        v16h b = *reinterpret_cast<const v16h*>(bp);

        acc = __builtin_amdgcn_wmma_f32_16x16x32_f16(
            /*neg_a=*/false, a, /*neg_b=*/false, b,
            /*c_mod=*/(short)0, acc, /*reuse_a=*/false, /*reuse_b=*/false);
    }

    // ---- store D and accumulate BN statistics ----
    const int col = ct * 16 + nloc;
    float s = 0.0f, q = 0.0f;
    #pragma unroll
    for (int r = 0; r < 8; ++r) {
        int m = rowBase + hi * 8 + r;
        float v = acc[r];
        hout[(long long)m * Cout + col] = v;
        s += v;
        q += v * v;
    }
    // combine the two hi-halves (same column set) then one atomic per column
    s += __shfl_xor(s, 16, 32);
    q += __shfl_xor(q, 16, 32);
    if (hi == 0) {
        atomicAdd(&sumOut[col], s);
        atomicAdd(&sqOut[col], q);
    }
}

// ---------------------------------------------------------------------------
// BatchNorm training-mode params from batch statistics
// ---------------------------------------------------------------------------
__global__ void bn_params(const float* __restrict__ sum, const float* __restrict__ sq,
                          const float* __restrict__ gamma, const float* __restrict__ beta,
                          float* __restrict__ scale, float* __restrict__ shift,
                          int C, float invN) {
    int c = threadIdx.x;
    if (c < C) {
        float mu  = sum[c] * invN;
        float var = sq[c] * invN - mu * mu;
        float sc  = gamma[c] * rsqrtf(var + EPS_BN);
        scale[c] = sc;
        shift[c] = beta[c] - mu * sc;
    }
}

// ---------------------------------------------------------------------------
// Head: gather per-point features (BN3+ReLU fused), per-batch channel sums
// ---------------------------------------------------------------------------
__global__ __launch_bounds__(256)
void head_accum(const float* __restrict__ h3,      // [N,16] raw conv3 out
                const int*   __restrict__ pcid,    // [bs*P]
                const float* __restrict__ scale,   // [16]
                const float* __restrict__ shift,   // [16]
                float* __restrict__ acc,           // [bs*16]
                int P) {
    __shared__ float sacc[16];
    __shared__ float sbn[32];
    int tid = threadIdx.x;
    if (tid < 16) { sacc[tid] = 0.0f; sbn[tid] = scale[tid]; sbn[16 + tid] = shift[tid]; }
    __syncthreads();

    int b = blockIdx.y;
    float local[16];
    #pragma unroll
    for (int c = 0; c < 16; ++c) local[c] = 0.0f;

    for (int p = blockIdx.x * blockDim.x + tid; p < P; p += gridDim.x * blockDim.x) {
        int v = pcid[(long long)b * P + p];
        const float* f = h3 + (long long)v * 16;
        #pragma unroll
        for (int c = 0; c < 16; ++c)
            local[c] += fmaxf(fmaf(f[c], sbn[c], sbn[16 + c]), 0.0f);
    }
    #pragma unroll
    for (int c = 0; c < 16; ++c) atomicAdd(&sacc[c], local[c]);
    __syncthreads();
    if (tid < 16) atomicAdd(&acc[b * 16 + tid], sacc[tid]);
}

// ---------------------------------------------------------------------------
// Final: out[s][b][j]; mean commutes with affine head:
//   mean(pcf @ W + bias) = (sum(pcf)/P) @ W + bias
// stack order: [green_1, green_2, red_1, red_2]
// ---------------------------------------------------------------------------
__global__ void final_out(const float* __restrict__ acc,  // [2][bs][16]
                          const float* __restrict__ Wg, const float* __restrict__ bg,
                          const float* __restrict__ Wr, const float* __restrict__ br,
                          float* __restrict__ out, int bs, float invP) {
    int t = threadIdx.x;
    if (t < 4 * bs * 3) {
        int s = t / (bs * 3);
        int b = (t / 3) % bs;
        int j = t % 3;
        int branch = s & 1;                   // 0:g1 1:g2 2:r1 3:r2
        const float* W    = (s < 2) ? Wg : Wr;
        const float* bias = (s < 2) ? bg : br;
        const float* a    = acc + (branch * bs + b) * 16;
        float r = 0.0f;
        #pragma unroll
        for (int c = 0; c < 16; ++c) r += a[c] * W[c * 3 + j];
        out[t] = r * invP + bias[j];
    }
}

// ---------------------------------------------------------------------------
// launch
// ---------------------------------------------------------------------------
static void run_branch(const float* vf, const int* nbr, const int* pcid,
                       const float* g1, const float* b1,
                       const float* g2, const float* b2,
                       const float* g3, const float* b3,
                       const _Float16* Bp1, const _Float16* Bp2, const _Float16* Bp3,
                       float* h1, float* h2, float* h3, _Float16* xh,
                       float* stats, float* acc,
                       int N, int P, hipStream_t stream) {
    // stats region layout (floats):
    float* sum1 = stats +   0; float* sq1 = stats +  16; float* sc1 = stats +  32; float* sh1 = stats +  48;
    float* sum2 = stats +  64; float* sq2 = stats +  96; float* sc2 = stats + 128; float* sh2 = stats + 160;
    float* sum3 = stats + 192; float* sq3 = stats + 208; float* sc3 = stats + 224; float* sh3 = stats + 240;

    const int tiles = (N + 15) / 16;
    const float invN = 1.0f / (float)N;

    // layer 1: Cin=6 (pad 8), Cout=16, K=224
    {
        int np = N * 8;
        pack_f16<6, 8, false><<<(np + 255) / 256, 256, 0, stream>>>(vf, nullptr, nullptr, xh, N);
        int blocks = (tiles * 1 + 3) / 4;
        conv_wmma<8, 16, 224><<<blocks, 128, 0, stream>>>(xh, nbr, Bp1, h1, sum1, sq1, N);
        bn_params<<<1, 64, 0, stream>>>(sum1, sq1, g1, b1, sc1, sh1, 16, invN);
    }
    // layer 2: Cin=16, Cout=32, K=448
    {
        int np = N * 16;
        pack_f16<16, 16, true><<<(np + 255) / 256, 256, 0, stream>>>(h1, sc1, sh1, xh, N);
        int blocks = (tiles * 2 + 3) / 4;
        conv_wmma<16, 32, 448><<<blocks, 128, 0, stream>>>(xh, nbr, Bp2, h2, sum2, sq2, N);
        bn_params<<<1, 64, 0, stream>>>(sum2, sq2, g2, b2, sc2, sh2, 32, invN);
    }
    // layer 3: Cin=32, Cout=16, K=864
    {
        int np = N * 32;
        pack_f16<32, 32, true><<<(np + 255) / 256, 256, 0, stream>>>(h2, sc2, sh2, xh, N);
        int blocks = (tiles * 1 + 3) / 4;
        conv_wmma<32, 16, 864><<<blocks, 128, 0, stream>>>(xh, nbr, Bp3, h3, sum3, sq3, N);
        bn_params<<<1, 64, 0, stream>>>(sum3, sq3, g3, b3, sc3, sh3, 16, invN);
    }
    // head accumulation
    {
        dim3 grid(64, 4);
        head_accum<<<grid, 256, 0, stream>>>(h3, pcid, sc3, sh3, acc, P);
    }
}

extern "C" void kernel_launch(void* const* d_in, const int* in_sizes, int n_in,
                              void* d_out, int out_size, void* d_ws, size_t ws_size,
                              hipStream_t stream) {
    const float* vf1   = (const float*)d_in[0];
    const int*   nbr1  = (const int*)  d_in[1];
    const int*   pcid1 = (const int*)  d_in[2];
    const float* vf2   = (const float*)d_in[3];
    const int*   nbr2  = (const int*)  d_in[4];
    const int*   pcid2 = (const int*)  d_in[5];
    const float* W1 = (const float*)d_in[7];
    const float* g1 = (const float*)d_in[8];
    const float* b1 = (const float*)d_in[9];
    const float* W2 = (const float*)d_in[10];
    const float* g2 = (const float*)d_in[11];
    const float* b2 = (const float*)d_in[12];
    const float* W3 = (const float*)d_in[13];
    const float* g3 = (const float*)d_in[14];
    const float* b3 = (const float*)d_in[15];
    const float* Wg = (const float*)d_in[16];
    const float* bg = (const float*)d_in[17];
    const float* Wr = (const float*)d_in[18];
    const float* br = (const float*)d_in[19];

    const int N  = in_sizes[0] / 6;          // 120000
    const int bs = out_size / 12;            // out is [4, bs, 3]
    const int P  = in_sizes[2] / bs;         // 60000

    // workspace layout (floats; all offsets multiples of 16 floats -> 64B aligned)
    float* ws   = (float*)d_ws;
    size_t off  = 0;
    float* h1    = ws + off; off += (size_t)N * 16;
    float* h2    = ws + off; off += (size_t)N * 32;
    float* h3    = ws + off; off += (size_t)N * 16;
    float* stat1 = ws + off; off += 256;
    float* stat2 = ws + off; off += 256;
    float* acc   = ws + off; off += 128;            // [branch][bs][16]
    _Float16* xh  = (_Float16*)(ws + off); off += (size_t)N * 16;  // N*32 f16
    _Float16* Bp1 = (_Float16*)(ws + off); off += 7 * 512 / 2;     // 3584 f16
    _Float16* Bp2 = (_Float16*)(ws + off); off += 28 * 512 / 2;    // 14336 f16
    _Float16* Bp3 = (_Float16*)(ws + off); off += 27 * 512 / 2;    // 13824 f16

    // zero stats + accumulators (deterministic per launch)
    {
        int nz = 512 + 128;
        zero_f32<<<(nz + 255) / 256, 256, 0, stream>>>(stat1, nz);
    }

    // pack weight fragments once (shared by both branches)
    pack_B<6, 8, 16, 224><<<(7 * 512 + 255) / 256, 256, 0, stream>>>(W1, Bp1);
    pack_B<16, 16, 32, 448><<<(28 * 512 + 255) / 256, 256, 0, stream>>>(W2, Bp2);
    pack_B<32, 32, 16, 864><<<(27 * 512 + 255) / 256, 256, 0, stream>>>(W3, Bp3);

    run_branch(vf1, nbr1, pcid1, g1, b1, g2, b2, g3, b3, Bp1, Bp2, Bp3,
               h1, h2, h3, xh, stat1, acc + 0, N, P, stream);
    run_branch(vf2, nbr2, pcid2, g1, b1, g2, b2, g3, b3, Bp1, Bp2, Bp3,
               h1, h2, h3, xh, stat2, acc + bs * 16, N, P, stream);

    final_out<<<1, 64, 0, stream>>>(acc, Wg, bg, Wr, br, (float*)d_out,
                                    bs, 1.0f / (float)P);
}
